// CausalFunctor_26594437497330
// MI455X (gfx1250) — compile-verified, matrix operand order checked by reading, not run on hardware
//
#include <hip/hip_runtime.h>

// ---------------------------------------------------------------------------
// Problem constants (reference: B=4, T=4096, D=1024)
// ---------------------------------------------------------------------------
#define CB   4
#define CT   4096
#define CD   1024
#define CM   (CB * CT)        // 16384 rows
#define CN2  (2 * CD)         // 2048
#define NC   16               // scan chunks along T
#define CL   (CT / NC)        // 256 steps per chunk

typedef __attribute__((ext_vector_type(16))) __bf16        v16bf;
typedef __attribute__((ext_vector_type(8)))  float         v8f;
typedef __attribute__((ext_vector_type(4)))  unsigned int  u32x4;
typedef __attribute__((ext_vector_type(4)))  int           v4i;

#define AS1 __attribute__((address_space(1)))
#define AS3 __attribute__((address_space(3)))

// CDNA5 async global->LDS copy path (direct-to-LDS, tracked by ASYNCcnt)
#if defined(__has_builtin)
#  if __has_builtin(__builtin_amdgcn_global_load_async_to_lds_b128)
#    define HAVE_ASYNC_COPY 1
#  endif
#endif
#ifndef HAVE_ASYNC_COPY
#  define HAVE_ASYNC_COPY 0
#endif

__device__ __forceinline__ void async_copy_wait0() {
#if HAVE_ASYNC_COPY
#  if __has_builtin(__builtin_amdgcn_s_wait_asynccnt)
    __builtin_amdgcn_s_wait_asynccnt(0);
#  else
    asm volatile("s_wait_asynccnt 0" ::: "memory");
#  endif
#endif
}

// one 16-byte global->LDS stage (async if available, else sync through VGPRs)
__device__ __forceinline__ void stage16B(const unsigned short* gp, unsigned short* lp) {
#if HAVE_ASYNC_COPY
    __builtin_amdgcn_global_load_async_to_lds_b128((AS1 v4i*)gp, (AS3 v4i*)lp, 0, 0);
#else
    *(u32x4*)lp = *(const u32x4*)gp;
#endif
}

__device__ __forceinline__ float fast_rcp(float x) {
#if defined(__has_builtin) && __has_builtin(__builtin_amdgcn_rcpf)
    return __builtin_amdgcn_rcpf(x);
#else
    return 1.0f / x;
#endif
}

__device__ __forceinline__ float silu_fast(float v) {
    return v * fast_rcp(1.0f + __expf(-v));   // v_exp + v_add + v_rcp + v_mul
}

__device__ __forceinline__ unsigned short f32_to_bf16(float f) {
    unsigned int u = __float_as_uint(f);
    unsigned int r = u + 0x7FFFu + ((u >> 16) & 1u);   // round-to-nearest-even
    return (unsigned short)(r >> 16);
}

// ---------------------------------------------------------------------------
// f32 -> bf16 convert (x)
// ---------------------------------------------------------------------------
__global__ void k_cvt_bf16(const float* __restrict__ src,
                           unsigned short* __restrict__ dst, int n) {
    int i = blockIdx.x * blockDim.x + threadIdx.x;
    if (i < n) dst[i] = f32_to_bf16(src[i]);
}

// transpose + convert: src (rows x cols) f32  ->  dst (cols x rows) bf16
__global__ void k_cvt_bf16_t(const float* __restrict__ src,
                             unsigned short* __restrict__ dst,
                             int rows, int cols) {
    int i = blockIdx.x * blockDim.x + threadIdx.x;
    if (i >= rows * cols) return;
    int r = i / cols, c = i % cols;
    dst[(size_t)c * rows + r] = f32_to_bf16(src[i]);
}

// ---------------------------------------------------------------------------
// bf16 WMMA GEMM:  C(MxN) = A(MxK) * BT(NxK)^T   [BT holds B transposed]
// Block tile 128x128, K-step 32, 8 waves (4 along M x 2 along N),
// wave tile 32x64 = 2x4 fragments of 16x16. Double-buffered padded LDS,
// staged with GLOBAL_LOAD_ASYNC_TO_LDS_B128 when available.
// MODE 0: C = silu(AB) stored bf16.  MODE 1: C = AB + bias stored f32.
// ---------------------------------------------------------------------------
template <int MODE>
__global__ __launch_bounds__(256)
void k_gemm_bf16(const unsigned short* __restrict__ A,
                 const unsigned short* __restrict__ BT,
                 void* __restrict__ Cout,
                 const float* __restrict__ bias,
                 int M, int N, int K) {
    constexpr int BM = 128, BN = 128, BK = 32, PAD = 40; // 80-byte rows: no bank conflicts
    __shared__ unsigned short lA[2][BM * PAD];
    __shared__ unsigned short lB[2][BN * PAD];

    const int tid  = threadIdx.x;
    const int lane = tid & 31;
    const int wave = tid >> 5;
    const int wm   = wave & 3;            // wave row   (4 along M)
    const int wn   = wave >> 2;           // wave col   (2 along N)
    const int bm0  = blockIdx.y * BM;
    const int bn0  = blockIdx.x * BN;

    v8f acc[2][4];
    #pragma unroll
    for (int i = 0; i < 2; ++i)
        #pragma unroll
        for (int j = 0; j < 4; ++j) acc[i][j] = (v8f)0.0f;

    // tile staging: 256 threads x 2 iters x 16B covers one 128x32 bf16 tile
    const int lr = tid >> 2;              // 0..63  (row within half-tile)
    const int lc = (tid & 3) * 8;         // 0,8,16,24 (bf16 col offset)

    // prologue: stage K-tile 0 into buffer 0
    #pragma unroll
    for (int i = 0; i < 2; ++i) {
        int row = lr + i * 64;
        stage16B(A  + (size_t)(bm0 + row) * K + lc, &lA[0][row * PAD + lc]);
        stage16B(BT + (size_t)(bn0 + row) * K + lc, &lB[0][row * PAD + lc]);
    }
    async_copy_wait0();
    __syncthreads();

    int buf = 0;
    const int nk = K / BK;
    for (int kt = 0; kt < nk; ++kt) {
        // stage next K-tile into buf^1 (its previous readers all passed the
        // barrier that ended the prior iteration, so this is race-free)
        if (kt + 1 < nk) {
            int kn = (kt + 1) * BK;
            #pragma unroll
            for (int i = 0; i < 2; ++i) {
                int row = lr + i * 64;
                stage16B(A  + (size_t)(bm0 + row) * K + kn + lc,
                         &lA[buf ^ 1][row * PAD + lc]);
                stage16B(BT + (size_t)(bn0 + row) * K + kn + lc,
                         &lB[buf ^ 1][row * PAD + lc]);
            }
        }

        // ---- fragments from LDS (ISA 16-bit A/B layout: lane<16 -> K 0-7,16-23;
        //      lane>=16 -> K 8-15,24-31; two 16B chunks per lane) ----
        const int hs = (lane >> 4) * 8;          // bf16 element offset 0 / 8
        union Frag { v16bf v; u32x4 q[2]; };
        Frag af[2], bf[4];
        #pragma unroll
        for (int fm = 0; fm < 2; ++fm) {
            int row = wm * 32 + fm * 16 + (lane & 15);
            af[fm].q[0] = *(const u32x4*)&lA[buf][row * PAD + hs];
            af[fm].q[1] = *(const u32x4*)&lA[buf][row * PAD + 16 + hs];
        }
        #pragma unroll
        for (int fn = 0; fn < 4; ++fn) {
            int row = wn * 64 + fn * 16 + (lane & 15);
            bf[fn].q[0] = *(const u32x4*)&lB[buf][row * PAD + hs];
            bf[fn].q[1] = *(const u32x4*)&lB[buf][row * PAD + 16 + hs];
        }

        #pragma unroll
        for (int fm = 0; fm < 2; ++fm)
            #pragma unroll
            for (int fn = 0; fn < 4; ++fn)
                acc[fm][fn] = __builtin_amdgcn_wmma_f32_16x16x32_bf16(
                    false, af[fm].v, false, bf[fn].v,
                    (short)0, acc[fm][fn], false, false);

        if (kt + 1 < nk) {
            async_copy_wait0();     // our async writes to buf^1 landed in LDS
            __syncthreads();        // everyone's writes visible, reads of buf done
            buf ^= 1;
        }
    }

    // ---- epilogue: C/D layout -> element r of v8f is row r + 8*(lane>=16), col = lane&15
    #pragma unroll
    for (int fm = 0; fm < 2; ++fm) {
        int rbase = bm0 + wm * 32 + fm * 16 + ((lane >> 4) << 3);
        #pragma unroll
        for (int fn = 0; fn < 4; ++fn) {
            int col = bn0 + wn * 64 + fn * 16 + (lane & 15);
            #pragma unroll
            for (int r = 0; r < 8; ++r) {
                float v = acc[fm][fn][r];
                size_t idx = (size_t)(rbase + r) * N + col;
                if (MODE == 0) {
                    ((unsigned short*)Cout)[idx] = f32_to_bf16(silu_fast(v));
                } else {
                    ((float*)Cout)[idx] = v + bias[col];
                }
            }
        }
    }
}

// ---------------------------------------------------------------------------
// Gated scan  h_t = a*h_{t-1} + (1-a)*tanh(theta), 3-pass chunked scan.
// mp layout: (B,T,2D); theta = mp[...,d], alpha = mp[...,D+d]
// ---------------------------------------------------------------------------
__device__ __forceinline__ void gate_ab(float th, float al, float& a, float& bb) {
    a  = 1.0f / (1.0f + expf(-al));
    bb = (1.0f - a) * tanhf(th);
}

__global__ void k_scan_carry(const float* __restrict__ mp,
                             float* __restrict__ cA, float* __restrict__ cB) {
    int tid = blockIdx.x * blockDim.x + threadIdx.x;     // B*D*NC threads
    if (tid >= CB * CD * NC) return;
    const int b  = tid / (CD * NC);
    const int ch = (tid / CD) % NC;
    const int d  = tid % CD;
    const float* base = mp + ((size_t)b * CT + (size_t)ch * CL) * CN2;
    float A = 1.0f, Bc = 0.0f;
    for (int t = 0; t < CL; ++t) {
        float a, bb;
        gate_ab(base[(size_t)t * CN2 + d], base[(size_t)t * CN2 + CD + d], a, bb);
        A *= a; Bc = a * Bc + bb;
    }
    size_t ci = ((size_t)b * CD + d) * NC + ch;
    cA[ci] = A; cB[ci] = Bc;
}

__global__ void k_scan_combine(const float* __restrict__ cA,
                               const float* __restrict__ cB,
                               float* __restrict__ hE) {
    int tid = blockIdx.x * blockDim.x + threadIdx.x;     // B*D threads
    if (tid >= CB * CD) return;
    float h = 0.0f;
    #pragma unroll
    for (int c = 0; c < NC; ++c) {
        hE[(size_t)tid * NC + c] = h;                    // h entering chunk c
        h = cA[(size_t)tid * NC + c] * h + cB[(size_t)tid * NC + c];
    }
}

__global__ void k_scan_apply(const float* __restrict__ mp,
                             const float* __restrict__ hE,
                             float* __restrict__ sem) {
    int tid = blockIdx.x * blockDim.x + threadIdx.x;     // B*D*NC threads
    if (tid >= CB * CD * NC) return;
    const int b  = tid / (CD * NC);
    const int ch = (tid / CD) % NC;
    const int d  = tid % CD;
    const float* base = mp + ((size_t)b * CT + (size_t)ch * CL) * CN2;
    float h = hE[((size_t)b * CD + d) * NC + ch];
    for (int t = 0; t < CL; ++t) {
        float a, bb;
        gate_ab(base[(size_t)t * CN2 + d], base[(size_t)t * CN2 + CD + d], a, bb);
        h = a * h + bb;
        sem[((size_t)b * CT + (size_t)ch * CL + t) * CD + d] = h;
    }
}

// ---------------------------------------------------------------------------
// Fused: causal depthwise conv(k=3) + SiLU + residual + LayerNorm + L2 norm
// One block per (b,t); 256 threads x 4 channels.
// ---------------------------------------------------------------------------
__device__ __forceinline__ float block_sum(float v, float* red) {
    int tid = threadIdx.x;
    red[tid] = v; __syncthreads();
    #pragma unroll
    for (int off = 128; off > 0; off >>= 1) {
        if (tid < off) red[tid] += red[tid + off];
        __syncthreads();
    }
    float r = red[0]; __syncthreads();
    return r;
}

__global__ __launch_bounds__(256)
void k_finalize(const float* __restrict__ sem,
                const float* __restrict__ conv_w,
                const float* __restrict__ gamma,
                const float* __restrict__ beta,
                float* __restrict__ out) {
    __shared__ float red[256];
    const int bt  = blockIdx.x;
    const int b   = bt / CT;
    const int t   = bt % CT;
    const int tid = threadIdx.x;
    const float* sbase = sem + ((size_t)b * CT + t) * CD;

    float y[4], s = 0.0f, s2 = 0.0f;
    #pragma unroll
    for (int c = 0; c < 4; ++c) {
        int d = c * 256 + tid;
        float x2 = sbase[d];
        float x1 = (t >= 1) ? sbase[d - CD]     : 0.0f;
        float x0 = (t >= 2) ? sbase[d - 2 * CD] : 0.0f;
        float cb = conv_w[d * 3] * x0 + conv_w[d * 3 + 1] * x1 + conv_w[d * 3 + 2] * x2;
        float yy = x2 + 0.1f * silu_fast(cb);
        y[c] = yy; s += yy; s2 += yy * yy;
    }
    float S  = block_sum(s,  red);
    float S2 = block_sum(s2, red);
    float mu  = S  * (1.0f / CD);
    float var = S2 * (1.0f / CD) - mu * mu;
    float rs  = rsqrtf(var + 1e-5f);

    float z[4], n2 = 0.0f;
    #pragma unroll
    for (int c = 0; c < 4; ++c) {
        int d = c * 256 + tid;
        z[c] = (y[c] - mu) * rs * gamma[d] + beta[d];
        n2 += z[c] * z[c];
    }
    float N2  = block_sum(n2, red);
    float inv = 1.0f / fmaxf(sqrtf(N2), 1e-12f);

    float* obase = out + ((size_t)b * CT + t) * CD;
    #pragma unroll
    for (int c = 0; c < 4; ++c) {
        int d = c * 256 + tid;
        obase[d] = z[c] * inv;
    }
}

// ---------------------------------------------------------------------------
// Launcher
// ---------------------------------------------------------------------------
extern "C" void kernel_launch(void* const* d_in, const int* in_sizes, int n_in,
                              void* d_out, int out_size, void* d_ws, size_t ws_size,
                              hipStream_t stream) {
    (void)in_sizes; (void)n_in; (void)out_size;
    const float* x      = (const float*)d_in[0];   // (B,T,D)
    const float* W1     = (const float*)d_in[1];   // (D, 2D)
    const float* W2     = (const float*)d_in[2];   // (2D, 2D)
    const float* b2     = (const float*)d_in[3];   // (2D,)
    const float* conv_w = (const float*)d_in[4];   // (D,1,3)
    const float* gamma  = (const float*)d_in[5];
    const float* beta   = (const float*)d_in[6];
    float* out = (float*)d_out;

    // workspace carve-up (256B aligned)
    auto align = [](size_t v) { return (v + 255) & ~(size_t)255; };
    size_t off = 0;
    size_t o_xb  = off; off += align((size_t)CM  * CD  * 2);  // x  bf16
    size_t o_w1t = off; off += align((size_t)CN2 * CD  * 2);  // W1^T bf16 (2048x1024)
    size_t o_w2t = off; off += align((size_t)CN2 * CN2 * 2);  // W2^T bf16 (2048x2048)
    size_t o_H   = off; off += align((size_t)CM  * CN2 * 2);  // silu(xW1) bf16
    size_t o_mp  = off; off += align((size_t)CM  * CN2 * 4);  // mp f32
    size_t o_sem = off; off += align((size_t)CM  * CD  * 4);  // semantic f32
    size_t o_cA  = off; off += align((size_t)CB * CD * NC * 4);
    size_t o_cB  = off; off += align((size_t)CB * CD * NC * 4);
    size_t o_hE  = off; off += align((size_t)CB * CD * NC * 4);
    if (ws_size < off) return;  // deterministic no-op if scratch too small

    char* ws = (char*)d_ws;
    unsigned short* xb  = (unsigned short*)(ws + o_xb);
    unsigned short* w1t = (unsigned short*)(ws + o_w1t);
    unsigned short* w2t = (unsigned short*)(ws + o_w2t);
    unsigned short* H   = (unsigned short*)(ws + o_H);
    float* mp  = (float*)(ws + o_mp);
    float* sem = (float*)(ws + o_sem);
    float* cA  = (float*)(ws + o_cA);
    float* cB  = (float*)(ws + o_cB);
    float* hE  = (float*)(ws + o_hE);

    // 1) precision staging
    {
        int n = CM * CD;
        k_cvt_bf16<<<(n + 255) / 256, 256, 0, stream>>>(x, xb, n);
    }
    k_cvt_bf16_t<<<(CD * CN2 + 255) / 256, 256, 0, stream>>>(W1, w1t, CD, CN2);
    k_cvt_bf16_t<<<(CN2 * CN2 + 255) / 256, 256, 0, stream>>>(W2, w2t, CN2, CN2);

    // 2) GEMM1: H = silu(x @ W1), bf16 out
    dim3 g1(CN2 / 128, CM / 128);
    k_gemm_bf16<0><<<g1, 256, 0, stream>>>(xb, w1t, (void*)H, nullptr, CM, CN2, CD);

    // 3) GEMM2: mp = H @ W2 + b2, f32 out
    dim3 g2(CN2 / 128, CM / 128);
    k_gemm_bf16<1><<<g2, 256, 0, stream>>>(H, w2t, (void*)mp, b2, CM, CN2, CN2);

    // 4) gated scan (3-pass chunked)
    {
        int n1 = CB * CD * NC;
        k_scan_carry  <<<(n1 + 255) / 256, 256, 0, stream>>>(mp, cA, cB);
        int n2 = CB * CD;
        k_scan_combine<<<(n2 + 255) / 256, 256, 0, stream>>>(cA, cB, hE);
        k_scan_apply  <<<(n1 + 255) / 256, 256, 0, stream>>>(mp, hE, sem);
    }

    // 5) conv + silu + residual + LayerNorm + L2 normalize
    k_finalize<<<CB * CT, 256, 0, stream>>>(sem, conv_w, gamma, beta, out);
}
